// Efficient_NeRF_55379308314650
// MI455X (gfx1250) — matrix operand order repaired and checked by reading, not compiled, not run on hardware
//
#include <hip/hip_runtime.h>
#include <hip/hip_bf16.h>
#include <math.h>
#include <stdint.h>

// ---------------- problem constants ----------------
#define NN      1024
#define CC      2
#define SC_     64
#define SS_     4
#define SF_     256
#define GG      128
#define NEAR_   2.0f
#define FAR_    6.0f
#define XYZ_MIN_ (-1.2f)
#define XYZ_MAX_ (1.2f)
#define WIDTH_  256
#define K_C_    9
#define K_F_    16
#define BETA_   0.1f
#define WTHRESH_ 1e-4f
#define DIN_    72     // 3 + 6*10 + (1 + 2*4)
#define KPAD0   96     // DIN padded to multiple of 32

typedef _Float16 h16;
typedef __attribute__((ext_vector_type(16))) _Float16 v16h;
typedef __attribute__((ext_vector_type(8)))  _Float16 v8h;
typedef __attribute__((ext_vector_type(8)))  float    v8f;
typedef __attribute__((ext_vector_type(4)))  unsigned int u32x4;
typedef __attribute__((ext_vector_type(8)))  int i32x8;
typedef __attribute__((ext_vector_type(4)))  int i32x4;

#if defined(__has_builtin)
#if __has_builtin(__builtin_amdgcn_tensor_load_to_lds) && __has_builtin(__builtin_amdgcn_s_wait_tensorcnt)
#define HAVE_TDM 1
#endif
#endif

// ---------------- weight conversion ----------------
// W is [K][Nc] row-major f32 -> Wt is [Nc][KPAD] f16 (transposed, zero padded K)
__global__ __launch_bounds__(256) void conv_wt(const float* __restrict__ W,
                                               h16* __restrict__ Wt,
                                               int K, int Nc, int KPAD) {
  int idx = blockIdx.x * 256 + threadIdx.x;
  int total = Nc * KPAD;
  if (idx >= total) return;
  int n = idx / KPAD, k = idx % KPAD;
  Wt[idx] = (k < K) ? (h16)W[k * Nc + n] : (h16)0.0f;
}

// Pack head: col0 = Wsig ([K][1]), cols 1..SHW = Wsh ([K][SHW]) -> Wht [NH][K] f16
__global__ __launch_bounds__(256) void conv_head(const float* __restrict__ Wsig,
                                                 const float* __restrict__ Wsh,
                                                 const float* __restrict__ bsig,
                                                 const float* __restrict__ bsh,
                                                 h16* __restrict__ Wht,
                                                 float* __restrict__ bh,
                                                 int K, int NH, int SHW) {
  int idx = blockIdx.x * 256 + threadIdx.x;
  if (idx < NH * K) {
    int n = idx / K, k = idx % K;
    float v = (n == 0) ? Wsig[k] : ((n - 1) < SHW ? Wsh[k * SHW + (n - 1)] : 0.0f);
    Wht[idx] = (h16)v;
  }
  if (idx < NH) {
    bh[idx] = (idx == 0) ? bsig[0] : ((idx - 1) < SHW ? bsh[idx - 1] : 0.0f);
  }
}

__global__ void init_wmax(unsigned* wmax) {
  if (threadIdx.x < CC) wmax[threadIdx.x] = 0u;
}

// ---------------- TDM weight staging ----------------
// Stage one k-slice (32 contiguous k for Ncols rows) of Wt [Ncols][K] f16
// into LDS as packed [Ncols][32] f16, using the Tensor Data Mover.
#ifdef HAVE_TDM
__device__ inline void tdm_issue_slice(const h16* g, unsigned ldsOff,
                                       int strideElems, int ncols) {
  unsigned long long ga = (unsigned long long)(uintptr_t)g;
  u32x4 g0;
  g0[0] = 1u;                                   // count=1, user descriptor
  g0[1] = ldsOff;                               // lds_addr (bytes)
  g0[2] = (unsigned)ga;                         // global_addr[31:0]
  g0[3] = (unsigned)((ga >> 32) & 0x01ffffffull) | 0x80000000u; // ga[56:32] | type=2
  i32x8 g1;
  g1[0] = 1 << 16;                              // workgroup_mask=0, data_size=1 (2B)
  g1[1] = (int)(32u << 16);                     // tensor_dim0 = 32 (lo16)
  g1[2] = (int)(((unsigned)ncols & 0xffffu) << 16); // tensor_dim1 = ncols (lo16)
  g1[3] = (int)(32u << 16);                     // tile_dim0 = 32
  g1[4] = (int)((unsigned)ncols & 0xffffu);     // tile_dim1 = ncols, tile_dim2 = 0
  g1[5] = strideElems;                          // tensor_dim0_stride (elements)
  g1[6] = 0;
  g1[7] = 0;
  i32x4 z4 = {0, 0, 0, 0};
#if __clang_major__ >= 23
  i32x8 z8 = {0, 0, 0, 0, 0, 0, 0, 0};
  __builtin_amdgcn_tensor_load_to_lds(g0, g1, z4, z4, z8, 0);
#else
  __builtin_amdgcn_tensor_load_to_lds(g0, g1, z4, z4, 0);
#endif
}
#endif

__device__ inline void stage_issue(const h16* __restrict__ Wt, int slice, int K,
                                   int ncols, h16* dst, int wave, int tid) {
#ifdef HAVE_TDM
  (void)tid;
  if (wave == 0)
    tdm_issue_slice(Wt + slice * 32, (unsigned)(uintptr_t)(void*)dst, K, ncols);
#else
  (void)wave;
  const h16* src = Wt + slice * 32;
  for (int idx = tid; idx < ncols * 32; idx += 256) {
    int n = idx >> 5, k = idx & 31;
    dst[idx] = src[n * K + k];
  }
#endif
}

__device__ inline void stage_wait(int wave) {
#ifdef HAVE_TDM
  if (wave == 0) __builtin_amdgcn_s_wait_tensorcnt(0);
#else
  (void)wave;
#endif
}

// ---------------- WMMA layer primitives ----------------
// hin: LDS activations [32][K] f16, Wt: global [256][K] f16 (transposed),
// wbuf: LDS double buffer 2 x [256][32] f16.  Block = 8 waves: 2(M)x4(N) tiling,
// each wave -> 16 rows x 64 cols = 4 accumulators.  TDM double-buffer pipeline:
// issue slice i+1, WMMA on slice i, s_wait_tensorcnt, barrier.
__device__ inline void mlp_layer(const h16* hin, int K,
                                 const h16* __restrict__ Wt,
                                 const float* __restrict__ bias,
                                 h16* hout, h16* wbuf,
                                 int wave, int lane, int tid) {
  const int wm = wave & 1, wn = wave >> 1;
  const int mbase = wm * 16, nbase = wn * 64;
  v8f zero = {0.f,0.f,0.f,0.f,0.f,0.f,0.f,0.f};
  v8f acc[4] = {zero, zero, zero, zero};
  const int mrow  = mbase + (lane & 15);
  const int aoff0 = (lane < 16) ? 0 : 16;
  const int aoff1 = (lane < 16) ? 8 : 24;
  const int nlane = lane & 15;
  const int koffB = (lane < 16) ? 0 : 16;
  const int nsl = K >> 5;

  stage_issue(Wt, 0, K, 256, wbuf, wave, tid);
  stage_wait(wave);
  __syncthreads();
  for (int i = 0; i < nsl; ++i) {
    const h16* wb = wbuf + (i & 1) * (256 * 32);
    if (i + 1 < nsl)
      stage_issue(Wt, i + 1, K, 256, wbuf + ((i + 1) & 1) * (256 * 32), wave, tid);
    const int ks = i * 32;
    v8h alo = *(const v8h*)(hin + mrow * K + ks + aoff0);
    v8h ahi = *(const v8h*)(hin + mrow * K + ks + aoff1);
    v16h a;
#pragma unroll
    for (int j = 0; j < 8; j++) { a[j] = alo[j]; a[8 + j] = ahi[j]; }
#pragma unroll
    for (int t = 0; t < 4; t++) {
      const h16* bp = wb + (nbase + t * 16 + nlane) * 32 + koffB;
      v16h b = *(const v16h*)bp;
      acc[t] = __builtin_amdgcn_wmma_f32_16x16x32_f16(
          false, a, false, b, (short)0, acc[t], false, false);
    }
    stage_wait(wave);
    __syncthreads();
  }
  // C/D layout: VGPR r -> row = r + (lane<16 ? 0 : 8), col = lane&15
  const int mw = mbase + ((lane < 16) ? 0 : 8);
#pragma unroll
  for (int t = 0; t < 4; t++) {
    int n = nbase + t * 16 + nlane;
    float bv = bias[n];
#pragma unroll
    for (int r = 0; r < 8; r++) {
      float v = acc[t][r] + bv;
      hout[(mw + r) * 256 + n] = (h16)(v > 0.f ? v : 0.f);  // ReLU
    }
  }
}

// Head: hin LDS [32][256], Wht [NH][256], out f32 LDS [32][NH] (bias, no relu).
// All waves participate in staging/barriers; only NH/8 waves compute.
template <int NH>
__device__ inline void mlp_head(const h16* hin,
                                const h16* __restrict__ Wht,
                                const float* __restrict__ bh,
                                float* outLDS, h16* wbuf,
                                int wave, int lane, int tid) {
  const bool active = (wave < NH / 8);  // 2(M) x NH/16(N) tiles
  const int wm = wave & 1, tn = wave >> 1;
  const int mbase = wm * 16, nbase = tn * 16;
  v8f acc = {0.f,0.f,0.f,0.f,0.f,0.f,0.f,0.f};
  const int mrow  = mbase + (lane & 15);
  const int aoff0 = (lane < 16) ? 0 : 16;
  const int aoff1 = (lane < 16) ? 8 : 24;
  const int nlane = lane & 15;
  const int koffB = (lane < 16) ? 0 : 16;

  stage_issue(Wht, 0, 256, NH, wbuf, wave, tid);
  stage_wait(wave);
  __syncthreads();
  for (int i = 0; i < 8; ++i) {
    const h16* wb = wbuf + (i & 1) * (256 * 32);
    if (i + 1 < 8)
      stage_issue(Wht, i + 1, 256, NH, wbuf + ((i + 1) & 1) * (256 * 32), wave, tid);
    if (active) {
      const int ks = i * 32;
      v8h alo = *(const v8h*)(hin + mrow * 256 + ks + aoff0);
      v8h ahi = *(const v8h*)(hin + mrow * 256 + ks + aoff1);
      v16h a;
#pragma unroll
      for (int j = 0; j < 8; j++) { a[j] = alo[j]; a[8 + j] = ahi[j]; }
      const h16* bp = wb + (nbase + nlane) * 32 + koffB;
      v16h b = *(const v16h*)bp;
      acc = __builtin_amdgcn_wmma_f32_16x16x32_f16(
          false, a, false, b, (short)0, acc, false, false);
    }
    stage_wait(wave);
    __syncthreads();
  }
  if (active) {
    const int mw = mbase + ((lane < 16) ? 0 : 8);
    const int n = nbase + nlane;
    float bv = bh[n];
#pragma unroll
    for (int r = 0; r < 8; r++)
      outLDS[(mw + r) * NH + n] = acc[r] + bv;
  }
}

// ---------------- features & SH ----------------
__device__ inline float feat_val(int j, const float* rd /* xyz0..2, cls */) {
  if (j < 3)  return rd[j];
  if (j < 33) { int t = j - 3;  int l = t / 3, d = t % 3; return sinf(rd[d] * (float)(1 << l)); }
  if (j < 63) { int t = j - 33; int l = t / 3, d = t % 3; return cosf(rd[d] * (float)(1 << l)); }
  if (j == 63) return rd[3];
  if (j < 68) return sinf(rd[3] * (float)(1 << (j - 64)));
  if (j < 72) return cosf(rd[3] * (float)(1 << (j - 68)));
  return 0.0f;
}

template <bool FINE>
__device__ inline float eval_sh_dev(const float* sh, float x, float y, float z) {
  float res = 0.28209479177387814f * sh[0];
  res += -0.4886025119029199f * y * sh[1] + 0.4886025119029199f * z * sh[2]
         - 0.4886025119029199f * x * sh[3];
  float xx = x*x, yy = y*y, zz = z*z, xy = x*y, yz = y*z, xz = x*z;
  res += 1.0925484305920792f * xy * sh[4] - 1.0925484305920792f * yz * sh[5]
       + 0.31539156525252005f * (2.f*zz - xx - yy) * sh[6]
       - 1.0925484305920792f * xz * sh[7]
       + 0.5462742152960396f * (xx - yy) * sh[8];
  if (FINE) {
    res += -0.5900435899266435f * y * (3.f*xx - yy) * sh[9]
         + 2.890611442640554f  * xy * z * sh[10]
         - 0.4570457994644658f * y * (4.f*zz - xx - yy) * sh[11]
         + 0.3731763325901154f * z * (2.f*zz - 3.f*xx - 3.f*yy) * sh[12]
         - 0.4570457994644658f * x * (4.f*zz - xx - yy) * sh[13]
         + 1.445305721320277f  * z * (xx - yy) * sh[14]
         - 0.5900435899266435f * x * (xx - 3.f*yy) * sh[15];
  }
  return res;
}

// ---------------- fused MLP kernel ----------------
template <bool FINE>
__global__ __launch_bounds__(256) void nerf_mlp(
    const float* __restrict__ rays, const float* __restrict__ clssArr,
    const float* __restrict__ dzArr,
    const h16* __restrict__ W0t, const h16* __restrict__ W1t,
    const h16* __restrict__ W2t, const h16* __restrict__ Wht,
    const float* __restrict__ b0, const float* __restrict__ b1,
    const float* __restrict__ b2, const float* __restrict__ bh,
    float* __restrict__ sigmaOut, float* __restrict__ rgbOut) {
  constexpr int S   = FINE ? SF_ : SC_;
  constexpr int NH  = FINE ? 64 : 32;
  constexpr int KSH = FINE ? K_F_ : K_C_;

  __shared__ h16  sFeat[32 * KPAD0];
  __shared__ h16  sH0[32 * 256];
  __shared__ h16  sH1[32 * 256];
  __shared__ h16  sWbuf[2 * 256 * 32];   // TDM double buffer for weight k-slices
  __shared__ float sHead[32 * NH];
  __shared__ float sRow[32][8];

  const int tid = threadIdx.x;
  const int wave = tid >> 5, lane = tid & 31;
  const int tileBase = blockIdx.x * 32;

  if (!FINE) {  // per-class coarse weights; class constant per 32-row tile
    const int c = (tileBase / S) % CC;
    W0t += (size_t)c * 256 * KPAD0;
    W1t += (size_t)c * 256 * 256;
    W2t += (size_t)c * 256 * 256;
    Wht += (size_t)c * 32 * 256;
    b0 += c * 256; b1 += c * 256; b2 += c * 256; bh += c * 32;
  }

  if (tid < 32) {
    int m = tileBase + tid;
    int rid = m / S, s = m % S;
    float ox = rays[rid*6+0], oy = rays[rid*6+1], oz = rays[rid*6+2];
    float dx = rays[rid*6+3], dy = rays[rid*6+4], dzc = rays[rid*6+5];
    float jit = dzArr[rid] * ((FAR_ - NEAR_) / (float)SC_);
    float z = NEAR_ + (FAR_ - NEAR_) * ((float)s / (float)(S - 1)) + jit;
    sRow[tid][0] = ox + dx * z;
    sRow[tid][1] = oy + dy * z;
    sRow[tid][2] = oz + dzc * z;
    sRow[tid][3] = clssArr[rid];
    float inv = 1.0f / (sqrtf(dx*dx + dy*dy + dzc*dzc) + 1e-8f);
    sRow[tid][4] = dx * inv; sRow[tid][5] = dy * inv; sRow[tid][6] = dzc * inv;
    sRow[tid][7] = 0.f;
  }
  __syncthreads();
  for (int idx = tid; idx < 32 * KPAD0; idx += 256) {
    int r = idx / KPAD0, j = idx % KPAD0;
    sFeat[idx] = (h16)feat_val(j, sRow[r]);
  }
  __syncthreads();
  mlp_layer(sFeat, KPAD0, W0t, b0, sH0, sWbuf, wave, lane, tid);
  __syncthreads();
  mlp_layer(sH0, 256, W1t, b1, sH1, sWbuf, wave, lane, tid);
  __syncthreads();
  mlp_layer(sH1, 256, W2t, b2, sH0, sWbuf, wave, lane, tid);
  __syncthreads();
  mlp_head<NH>(sH0, Wht, bh, sHead, sWbuf, wave, lane, tid);
  __syncthreads();

  if (tid < 32) sigmaOut[tileBase + tid] = sHead[tid * NH + 0];
  if (tid < 96) {  // 32 rows x 3 channels: SH eval + sigmoid
    int r = tid / 3, ch = tid % 3;
    const float* sh = &sHead[r * NH + 1 + ch * KSH];
    float res = eval_sh_dev<FINE>(sh, sRow[r][4], sRow[r][5], sRow[r][6]);
    rgbOut[(size_t)(tileBase + r) * 3 + ch] = 1.0f / (1.0f + expf(-res));
  }
}

// ---------------- render kernels (one thread per ray) ----------------
__device__ inline int vox_idx(float p) {
  int v = (int)((p - XYZ_MIN_) * ((float)GG / (XYZ_MAX_ - XYZ_MIN_)));
  return v < 0 ? 0 : (v > GG - 1 ? GG - 1 : v);
}

__global__ __launch_bounds__(256) void coarse_render(
    const float* __restrict__ rays, const float* __restrict__ dzArr,
    const float* __restrict__ voxIn, const float* __restrict__ sigmaC,
    const float* __restrict__ rgbC, float* __restrict__ wC,
    unsigned* __restrict__ wmax, float* __restrict__ rgbCoarseOut,
    float* __restrict__ voxOut) {
  int rid = blockIdx.x * 256 + threadIdx.x;
  if (rid >= NN * CC) return;
  int c = rid % CC;
  float ox = rays[rid*6+0], oy = rays[rid*6+1], oz = rays[rid*6+2];
  float dx = rays[rid*6+3], dy = rays[rid*6+4], dzc = rays[rid*6+5];
  float jit = dzArr[rid] * ((FAR_ - NEAR_) / (float)SC_);
  const float dzs = (FAR_ - NEAR_) / (float)(SC_ - 1);
  float T = 1.f, r0 = 0.f, r1 = 0.f, r2 = 0.f, wsum = 0.f, wmx = 0.f;
  for (int s = 0; s < SC_; s++) {
    int m = rid * SC_ + s;
    float z = NEAR_ + dzs * (float)s + jit;
    int v0 = vox_idx(ox + dx * z), v1 = vox_idx(oy + dy * z), v2 = vox_idx(oz + dzc * z);
    size_t flat = (((size_t)(c * GG + v0) * GG + v1) * GG) + v2;
    float so = voxIn[flat];
    bool mask = so > 0.0f;
    float sg = mask ? sigmaC[m] : 0.0f;
    float delta = (s == SC_ - 1) ? 1e10f : dzs;
    float alpha = 1.0f - expf(-delta * fmaxf(sg, 0.0f));
    float w = alpha * T;
    T *= (1.0f - alpha + 1e-10f);
    r0 += w * rgbC[(size_t)m*3+0]; r1 += w * rgbC[(size_t)m*3+1]; r2 += w * rgbC[(size_t)m*3+2];
    wsum += w; wmx = fmaxf(wmx, w);
    wC[m] = w;
    voxOut[flat] = mask ? (1.0f - BETA_) * so + BETA_ * sg : so;
  }
  float bg = 1.0f - wsum;
  rgbCoarseOut[(size_t)rid*3+0] = r0 + bg;
  rgbCoarseOut[(size_t)rid*3+1] = r1 + bg;
  rgbCoarseOut[(size_t)rid*3+2] = r2 + bg;
  atomicMax(&wmax[c], __float_as_uint(wmx));  // w >= 0 -> uint order == float order
}

__global__ __launch_bounds__(256) void fine_render(
    const float* __restrict__ sigmaF, const float* __restrict__ rgbF,
    const float* __restrict__ wC, const unsigned* __restrict__ wmax,
    float* __restrict__ rgbFineOut) {
  int rid = blockIdx.x * 256 + threadIdx.x;
  if (rid >= NN * CC) return;
  int c = rid % CC;
  float thresh = fminf(WTHRESH_, __uint_as_float(wmax[c]));
  const float dzs = (FAR_ - NEAR_) / (float)(SF_ - 1);
  float T = 1.f, r0 = 0.f, r1 = 0.f, r2 = 0.f, wsum = 0.f;
  for (int s = 0; s < SF_; s++) {
    int m = rid * SF_ + s;
    bool mask = wC[rid * SC_ + (s / SS_)] >= thresh;
    float sg = mask ? sigmaF[m] : 0.0f;
    float delta = (s == SF_ - 1) ? 1e10f : dzs;
    float alpha = 1.0f - expf(-delta * fmaxf(sg, 0.0f));
    float w = alpha * T;
    T *= (1.0f - alpha + 1e-10f);
    r0 += w * rgbF[(size_t)m*3+0]; r1 += w * rgbF[(size_t)m*3+1]; r2 += w * rgbF[(size_t)m*3+2];
    wsum += w;
  }
  float bg = 1.0f - wsum;
  rgbFineOut[(size_t)rid*3+0] = r0 + bg;
  rgbFineOut[(size_t)rid*3+1] = r1 + bg;
  rgbFineOut[(size_t)rid*3+2] = r2 + bg;
}

// ---------------- host launch ----------------
extern "C" void kernel_launch(void* const* d_in, const int* in_sizes, int n_in,
                              void* d_out, int out_size, void* d_ws, size_t ws_size,
                              hipStream_t stream) {
  const float* rays  = (const float*)d_in[0];
  const float* clss  = (const float*)d_in[1];
  const float* dzArr = (const float*)d_in[2];
  const float* cW0   = (const float*)d_in[3];   // (C,72,256)
  const float* cb0   = (const float*)d_in[4];   // (C,256)
  const float* cW1   = (const float*)d_in[5];
  const float* cb1   = (const float*)d_in[6];
  const float* cW2   = (const float*)d_in[7];
  const float* cb2   = (const float*)d_in[8];
  const float* cWsig = (const float*)d_in[9];   // (C,256,1)
  const float* cbsig = (const float*)d_in[10];  // (C,1)
  const float* cWsh  = (const float*)d_in[11];  // (C,256,27)
  const float* cbsh  = (const float*)d_in[12];  // (C,27)
  const float* fW0   = (const float*)d_in[13];
  const float* fb0   = (const float*)d_in[14];
  const float* fW1   = (const float*)d_in[15];
  const float* fb1   = (const float*)d_in[16];
  const float* fW2   = (const float*)d_in[17];
  const float* fb2   = (const float*)d_in[18];
  const float* fWsig = (const float*)d_in[19];
  const float* fbsig = (const float*)d_in[20];
  const float* fWsh  = (const float*)d_in[21];  // (256,48)
  const float* fbsh  = (const float*)d_in[22];
  const float* voxIn = (const float*)d_in[23];  // (C,G,G,G)
  (void)in_sizes; (void)n_in; (void)out_size; (void)ws_size;

  const int MC = NN * CC * SC_;   // 131072
  const int MF = NN * CC * SF_;   // 524288

  char* ws = (char*)d_ws;
  size_t o = 0;
  auto take = [&](size_t bytes) { size_t r = o; o = (o + bytes + 255) & ~(size_t)255; return r; };
  h16* cW0t = (h16*)(ws + take((size_t)CC * 256 * KPAD0 * 2));
  h16* cW1t = (h16*)(ws + take((size_t)CC * 256 * 256 * 2));
  h16* cW2t = (h16*)(ws + take((size_t)CC * 256 * 256 * 2));
  h16* cWht = (h16*)(ws + take((size_t)CC * 32 * 256 * 2));
  h16* fW0t = (h16*)(ws + take((size_t)256 * KPAD0 * 2));
  h16* fW1t = (h16*)(ws + take((size_t)256 * 256 * 2));
  h16* fW2t = (h16*)(ws + take((size_t)256 * 256 * 2));
  h16* fWht = (h16*)(ws + take((size_t)64 * 256 * 2));
  float* cbh = (float*)(ws + take((size_t)CC * 32 * 4));
  float* fbh = (float*)(ws + take((size_t)64 * 4));
  unsigned* wmax = (unsigned*)(ws + take((size_t)CC * 4));
  float* sigC = (float*)(ws + take((size_t)MC * 4));
  float* rgbC = (float*)(ws + take((size_t)MC * 3 * 4));
  float* wCbuf = (float*)(ws + take((size_t)MC * 4));
  float* sigF = (float*)(ws + take((size_t)MF * 4));
  float* rgbF = (float*)(ws + take((size_t)MF * 3 * 4));

  float* out = (float*)d_out;
  float* rgbCoarseOut = out;                  // (N,C,3)
  float* rgbFineOut   = out + (size_t)NN * CC * 3;
  float* voxOut       = out + (size_t)2 * NN * CC * 3;

  // 1) convert weights to transposed/padded f16
  for (int c = 0; c < CC; c++) {
    conv_wt<<<(256 * KPAD0 + 255) / 256, 256, 0, stream>>>(
        cW0 + (size_t)c * DIN_ * 256, cW0t + (size_t)c * 256 * KPAD0, DIN_, 256, KPAD0);
    conv_wt<<<(256 * 256 + 255) / 256, 256, 0, stream>>>(
        cW1 + (size_t)c * 256 * 256, cW1t + (size_t)c * 256 * 256, 256, 256, 256);
    conv_wt<<<(256 * 256 + 255) / 256, 256, 0, stream>>>(
        cW2 + (size_t)c * 256 * 256, cW2t + (size_t)c * 256 * 256, 256, 256, 256);
    conv_head<<<(32 * 256 + 255) / 256, 256, 0, stream>>>(
        cWsig + (size_t)c * 256, cWsh + (size_t)c * 256 * 27,
        cbsig + c, cbsh + (size_t)c * 27,
        cWht + (size_t)c * 32 * 256, cbh + (size_t)c * 32, 256, 32, 27);
  }
  conv_wt<<<(256 * KPAD0 + 255) / 256, 256, 0, stream>>>(fW0, fW0t, DIN_, 256, KPAD0);
  conv_wt<<<(256 * 256 + 255) / 256, 256, 0, stream>>>(fW1, fW1t, 256, 256, 256);
  conv_wt<<<(256 * 256 + 255) / 256, 256, 0, stream>>>(fW2, fW2t, 256, 256, 256);
  conv_head<<<(64 * 256 + 255) / 256, 256, 0, stream>>>(
      fWsig, fWsh, fbsig, fbsh, fWht, fbh, 256, 64, 48);
  init_wmax<<<1, 32, 0, stream>>>(wmax);

  // 2) seed voxel output with input voxels (scatter overwrites)
  hipMemcpyAsync(voxOut, voxIn, (size_t)CC * GG * GG * GG * sizeof(float),
                 hipMemcpyDeviceToDevice, stream);

  // 3) fused WMMA MLPs with TDM weight staging
  nerf_mlp<false><<<MC / 32, 256, 0, stream>>>(
      rays, clss, dzArr, cW0t, cW1t, cW2t, cWht, cb0, cb1, cb2, cbh, sigC, rgbC);
  nerf_mlp<true><<<MF / 32, 256, 0, stream>>>(
      rays, clss, dzArr, fW0t, fW1t, fW2t, fWht, fb0, fb1, fb2, fbh, sigF, rgbF);

  // 4) volume rendering + voxel EMA update + per-class w_max, then fine pass
  coarse_render<<<(NN * CC + 255) / 256, 256, 0, stream>>>(
      rays, dzArr, voxIn, sigC, rgbC, wCbuf, wmax, rgbCoarseOut, voxOut);
  fine_render<<<(NN * CC + 255) / 256, 256, 0, stream>>>(
      sigF, rgbF, wCbuf, wmax, rgbFineOut);
}
